// LinearAttention_84482006712578
// MI455X (gfx1250) — compile-verified, matrix operand order checked by reading, not run on hardware
//
#include <hip/hip_runtime.h>

#define SEQ   4096
#define CH    512
#define NCOL  1024
#define BATCH 16
#define HEADS 8
#define HDIM  64

typedef __bf16 v16bf __attribute__((ext_vector_type(16)));
typedef __bf16 v8bf  __attribute__((ext_vector_type(8)));
typedef __bf16 v4bf  __attribute__((ext_vector_type(4)));
typedef float  v8f   __attribute__((ext_vector_type(8)));

// ---- helpers -------------------------------------------------------------

__device__ __forceinline__ v8f wmma_bf16(v16bf a, v16bf b, v8f c) {
  // v_wmma_f32_16x16x32_bf16  (8-arg form: neg_a, A, neg_b, B, c_mod, C, reuse_a, reuse_b)
  return __builtin_amdgcn_wmma_f32_16x16x32_bf16(false, a, false, b, (short)0, c, false, false);
}

// Split an fp32 value into bf16 hi + bf16 lo (residual) for 3-term emulated-fp32 GEMM.
__device__ __forceinline__ void split1(float f, __bf16& h, __bf16& l) {
  h = (__bf16)f;
  l = (__bf16)(f - (float)h);
}

// Load a 16-element K-fragment from an LDS row laid out [row][k], row stride LDT=40 bf16.
// Lane half selects K = {half*8 .. half*8+7} U {16+half*8 .. 16+half*8+7}.
__device__ __forceinline__ v16bf ldfrag(const __bf16* rowp, int half) {
  v8bf a = *(const v8bf*)(rowp + half * 8);
  v8bf b = *(const v8bf*)(rowp + 16 + half * 8);
  return __builtin_shufflevector(a, b, 0,1,2,3,4,5,6,7,8,9,10,11,12,13,14,15);
}

// Build a hi/lo fragment pair directly from contiguous fp32 global memory (row base p).
__device__ __forceinline__ void gfrag(const float* p, int half, v16bf& hi, v16bf& lo) {
  float f[16];
  #pragma unroll
  for (int j = 0; j < 2; j++) {
    float4 a = *(const float4*)(p + j * 16 + half * 8);
    float4 b = *(const float4*)(p + j * 16 + half * 8 + 4);
    f[j*8+0]=a.x; f[j*8+1]=a.y; f[j*8+2]=a.z; f[j*8+3]=a.w;
    f[j*8+4]=b.x; f[j*8+5]=b.y; f[j*8+6]=b.z; f[j*8+7]=b.w;
  }
  #pragma unroll
  for (int i = 0; i < 16; i++) {
    __bf16 h, l;
    split1(f[i], h, l);
    hi[i] = h; lo[i] = l;
  }
}

// ---- kernel 0: zero the k-sum accumulator --------------------------------

__global__ void la_zero(float* __restrict__ p, int n) {
  int i = blockIdx.x * 256 + threadIdx.x;
  if (i < n) p[i] = 0.f;
}

// ---- kernel 1: qk GEMM (M=65536,N=1024,K=512) + bias + elu+1 + RoPE ------
// Block tile 128x128, 8 waves (4M x 2N), per-wave 2x4 accum tiles of 16x16.
// Emulated-fp32 via bf16 hi/lo split: 3 WMMAs per tile per K-chunk of 32.

__global__ __launch_bounds__(256) void la_gemm_qk(
    const float* __restrict__ x, const float* __restrict__ w,
    const float* __restrict__ bias,
    float* __restrict__ q_act, float* __restrict__ q_rope,
    float* __restrict__ k_rope, float* __restrict__ ksum)
{
  constexpr int LDT = 40;  // padded row stride (bf16 elems) -> conflict-free b128 reads
  __shared__ __align__(16) __bf16 Ahi[128 * LDT];
  __shared__ __align__(16) __bf16 Alo[128 * LDT];
  __shared__ __align__(16) __bf16 Bhi[128 * LDT];  // stored transposed: [n][k]
  __shared__ __align__(16) __bf16 Blo[128 * LDT];
  __shared__ float colsum[128];

  const int t    = threadIdx.x;
  const int lane = t & 31;
  const int wid  = t >> 5;
  const int lm   = lane & 15;
  const int half = lane >> 4;
  const int wm   = wid >> 1;   // 0..3
  const int wn   = wid & 1;    // 0..1
  const int m0   = blockIdx.y * 128;
  const int n0   = blockIdx.x * 128;

  const v8f vzero = {0.f,0.f,0.f,0.f,0.f,0.f,0.f,0.f};
  v8f acc[8];
  #pragma unroll
  for (int i = 0; i < 8; i++) acc[i] = vzero;

  const int sr  = t >> 3;        // 0..31
  const int sc4 = (t & 7) * 4;   // 0,4,..,28

  for (int kk = 0; kk < CH; kk += 32) {
    // stage A (x tile, 128 rows x 32 k) as hi/lo bf16, [m][k]
    #pragma unroll
    for (int p = 0; p < 4; p++) {
      int row = sr + p * 32;
      float4 v = *(const float4*)(x + (size_t)(m0 + row) * CH + kk + sc4);
      float fv[4] = {v.x, v.y, v.z, v.w};
      v4bf h, l;
      #pragma unroll
      for (int j = 0; j < 4; j++) {
        __bf16 hh, ll;
        split1(fv[j], hh, ll);
        h[j] = hh; l[j] = ll;
      }
      *(v4bf*)(Ahi + row * LDT + sc4) = h;
      *(v4bf*)(Alo + row * LDT + sc4) = l;
    }
    // stage B (w tile, 32 k x 128 n) transposed to [n][k]
    #pragma unroll
    for (int p = 0; p < 4; p++) {
      int nc = sc4 + p * 32;
      float4 v = *(const float4*)(w + (size_t)(kk + sr) * NCOL + n0 + nc);
      float fv[4] = {v.x, v.y, v.z, v.w};
      #pragma unroll
      for (int j = 0; j < 4; j++) {
        __bf16 hh, ll;
        split1(fv[j], hh, ll);
        Bhi[(nc + j) * LDT + sr] = hh;
        Blo[(nc + j) * LDT + sr] = ll;
      }
    }
    __syncthreads();

    v16bf ah[2], al[2], bh[4], bl[4];
    #pragma unroll
    for (int tm = 0; tm < 2; tm++) {
      int row = wm * 32 + tm * 16 + lm;
      ah[tm] = ldfrag(Ahi + row * LDT, half);
      al[tm] = ldfrag(Alo + row * LDT, half);
    }
    #pragma unroll
    for (int tn = 0; tn < 4; tn++) {
      int col = wn * 64 + tn * 16 + lm;
      bh[tn] = ldfrag(Bhi + col * LDT, half);
      bl[tn] = ldfrag(Blo + col * LDT, half);
    }
    #pragma unroll
    for (int tm = 0; tm < 2; tm++)
      #pragma unroll
      for (int tn = 0; tn < 4; tn++) {
        int i = tm * 4 + tn;
        acc[i] = wmma_bf16(ah[tm], bh[tn], acc[i]);
        acc[i] = wmma_bf16(ah[tm], bl[tn], acc[i]);
        acc[i] = wmma_bf16(al[tm], bh[tn], acc[i]);
      }
    __syncthreads();
  }

  // ---- epilogue: bias, elu+1, RoPE, k column sums ----
  const bool is_q  = (n0 < CH);            // N blocks 0..3 => q, 4..7 => k
  const int  batch = m0 >> 12;             // 128-row tile never crosses a batch
  const int  nbase = m0 & (SEQ - 1);
  if (t < 128) colsum[t] = 0.f;
  __syncthreads();

  #pragma unroll
  for (int tm = 0; tm < 2; tm++) {
    #pragma unroll
    for (int tn = 0; tn < 4; tn++) {
      const int i   = tm * 4 + tn;
      const int col = n0 + wn * 64 + tn * 16 + lm;
      const int ch  = is_q ? col : (col - CH);
      const float bs = bias[col];
      const int pr   = ch >> 1;
      const bool even = (ch & 1) == 0;
      // theta = 10000^(-(pr mod 128)/128) = exp(-(pr&127) * ln(10000)/128)
      const float theta = __expf(-(float)(pr & 127) * (9.210340371976184f / 128.0f));
      float csum = 0.f;
      #pragma unroll
      for (int v = 0; v < 8; ++v) {
        const int nrow = nbase + wm * 32 + tm * 16 + half * 8 + v;
        float val = acc[i][v] + bs;
        float act = (val > 0.f) ? (val + 1.f) : __expf(val);   // elu(x)+1
        float partner = __shfl_xor(act, 1);                    // RoPE pair exchange
        float pos = (float)((pr < 128) ? (nrow >> 6) : (nrow & 63)); // y first half, x second
        float s, c;
        __sincosf(pos * theta, &s, &c);
        float rope = even ? (act * c - partner * s) : (act * c + partner * s);
        size_t off = ((size_t)batch * SEQ + nrow) * CH + ch;
        if (is_q) { q_act[off] = act; q_rope[off] = rope; }
        else      { k_rope[off] = rope; csum += act; }
      }
      if (!is_q) {
        csum += __shfl_xor(csum, 16);                          // fold lane halves (same column)
        if (half == 0) atomicAdd(&colsum[wn * 64 + tn * 16 + lm], csum);
      }
    }
  }
  __syncthreads();
  if (!is_q && t < 128)
    atomicAdd(&ksum[batch * CH + (n0 - CH) + t], colsum[t]);
}

// ---- kernel 2: kv = (k_rope*s)^T (v*s) per (b,h), 64x64, K=4096 ----------
// Stores kv transposed [b][h][e][d] so kernel 4 gets contiguous B fragments.

__global__ __launch_bounds__(256) void la_kv(
    const float* __restrict__ k_rope, const float* __restrict__ x,
    float* __restrict__ kvT)
{
  constexpr int LDT = 40;
  __shared__ __align__(16) __bf16 Khi[64 * LDT];
  __shared__ __align__(16) __bf16 Klo[64 * LDT];
  __shared__ __align__(16) __bf16 Vhi[64 * LDT];
  __shared__ __align__(16) __bf16 Vlo[64 * LDT];

  const int t = threadIdx.x, lane = t & 31, wid = t >> 5;
  const int lm = lane & 15, half = lane >> 4;
  const int wm = wid >> 1, wn = wid & 1;
  const int h = blockIdx.x, b = blockIdx.y;

  const v8f vzero = {0.f,0.f,0.f,0.f,0.f,0.f,0.f,0.f};
  v8f acc[2] = {vzero, vzero};

  const size_t base = ((size_t)b * SEQ) * CH + h * HDIM;
  const int sr = t >> 3, sc4 = (t & 7) * 4;

  for (int n0 = 0; n0 < SEQ; n0 += 32) {
    // stage 32(n) x 64(ch) tiles of k_rope and v(=x), transposed to [ch][n]
    #pragma unroll
    for (int p = 0; p < 2; p++) {
      int cc = sc4 + p * 32;
      const size_t roff = base + (size_t)(n0 + sr) * CH + cc;
      float4 kv4 = *(const float4*)(k_rope + roff);
      float4 xv4 = *(const float4*)(x + roff);
      float fk[4] = {kv4.x, kv4.y, kv4.z, kv4.w};
      float fx[4] = {xv4.x, xv4.y, xv4.z, xv4.w};
      #pragma unroll
      for (int j = 0; j < 4; j++) {
        __bf16 hh, ll;
        split1(fk[j], hh, ll);
        Khi[(cc + j) * LDT + sr] = hh;
        Klo[(cc + j) * LDT + sr] = ll;
        split1(fx[j], hh, ll);
        Vhi[(cc + j) * LDT + sr] = hh;
        Vlo[(cc + j) * LDT + sr] = ll;
      }
    }
    __syncthreads();
    v16bf ah = ldfrag(Khi + (wm * 16 + lm) * LDT, half);
    v16bf al = ldfrag(Klo + (wm * 16 + lm) * LDT, half);
    #pragma unroll
    for (int tn = 0; tn < 2; tn++) {
      int e = wn * 32 + tn * 16 + lm;
      v16bf bh = ldfrag(Vhi + e * LDT, half);
      v16bf bl = ldfrag(Vlo + e * LDT, half);
      acc[tn] = wmma_bf16(ah, bh, acc[tn]);
      acc[tn] = wmma_bf16(ah, bl, acc[tn]);
      acc[tn] = wmma_bf16(al, bh, acc[tn]);
    }
    __syncthreads();
  }
  const float sc = 1.0f / (float)SEQ;   // scale^2 = n^-1 folded here
  #pragma unroll
  for (int tn = 0; tn < 2; tn++) {
    int e = wn * 32 + tn * 16 + lm;
    #pragma unroll
    for (int v = 0; v < 8; v++) {
      int d = wm * 16 + half * 8 + v;
      kvT[((size_t)(b * HEADS + h) * HDIM + e) * HDIM + d] = acc[tn][v] * sc;
    }
  }
}

// ---- kernel 3: z = 1/(q_act . mean_k + 1e-6) -----------------------------

__global__ __launch_bounds__(256) void la_z(
    const float* __restrict__ q_act, const float* __restrict__ ksum,
    float* __restrict__ zbuf)
{
  const int idx = blockIdx.x * 256 + threadIdx.x;   // [b][h][n]
  const int nn = idx & (SEQ - 1);
  const int bh = idx >> 12;
  const int h  = bh & (HEADS - 1);
  const int b  = bh >> 3;
  const float* qp = q_act + ((size_t)(b * SEQ) + nn) * CH + h * HDIM;
  const float* kp = ksum + b * CH + h * HDIM;
  float acc = 0.f;
  #pragma unroll 8
  for (int d = 0; d < HDIM; d++) acc += qp[d] * kp[d];
  zbuf[idx] = 1.0f / (acc * (1.0f / (float)SEQ) + 1e-6f);
}

// ---- kernel 4: out = (q_rope @ kv) * z + lepe ----------------------------
// Per (b,h,128-row tile). A/B fragments loaded straight from global (contiguous).

__global__ __launch_bounds__(256) void la_out(
    const float* __restrict__ q_rope, const float* __restrict__ kvT,
    const float* __restrict__ zbuf, const float* __restrict__ x,
    const float* __restrict__ lw, const float* __restrict__ lb,
    float* __restrict__ out)
{
  const int t = threadIdx.x, lane = t & 31, wid = t >> 5;
  const int lm = lane & 15, half = lane >> 4;
  const int wm = wid >> 1, wn = wid & 1;
  const int nbase = blockIdx.x * 128;
  const int h = blockIdx.y, b = blockIdx.z;

  const v8f vzero = {0.f,0.f,0.f,0.f,0.f,0.f,0.f,0.f};
  v8f acc[4] = {vzero, vzero, vzero, vzero};

  #pragma unroll
  for (int kc = 0; kc < 2; kc++) {      // K = 64 in two 32-chunks
    v16bf ah[2], al[2], bh[2], bl[2];
    #pragma unroll
    for (int tm = 0; tm < 2; tm++) {
      int nrow = nbase + wm * 32 + tm * 16 + lm;
      gfrag(q_rope + ((size_t)(b * SEQ + nrow)) * CH + h * HDIM + kc * 32, half, ah[tm], al[tm]);
    }
    #pragma unroll
    for (int tn = 0; tn < 2; tn++) {
      int e = wn * 32 + tn * 16 + lm;
      gfrag(kvT + ((size_t)((b * HEADS + h) * HDIM + e)) * HDIM + kc * 32, half, bh[tn], bl[tn]);
    }
    #pragma unroll
    for (int tm = 0; tm < 2; tm++)
      #pragma unroll
      for (int tn = 0; tn < 2; tn++) {
        int i = tm * 2 + tn;
        acc[i] = wmma_bf16(ah[tm], bh[tn], acc[i]);
        acc[i] = wmma_bf16(ah[tm], bl[tn], acc[i]);
        acc[i] = wmma_bf16(al[tm], bh[tn], acc[i]);
      }
  }

  #pragma unroll
  for (int tm = 0; tm < 2; tm++) {
    #pragma unroll
    for (int tn = 0; tn < 2; tn++) {
      const int i  = tm * 2 + tn;
      const int e  = wn * 32 + tn * 16 + lm;
      const int ch = h * HDIM + e;
      float wgt[9];
      #pragma unroll
      for (int j = 0; j < 9; j++) wgt[j] = lw[j * CH + ch];
      const float bsv = lb[ch];
      #pragma unroll
      for (int v = 0; v < 8; v++) {
        const int nrow = nbase + wm * 32 + tm * 16 + half * 8 + v;
        const float zv = zbuf[((size_t)(b * HEADS + h)) * SEQ + nrow];
        float o = acc[i][v] * zv;
        const int y = nrow >> 6, xx = nrow & 63;
        float lep = bsv;
        #pragma unroll
        for (int dy = 0; dy < 3; dy++) {
          int yy = y + dy - 1;
          if (yy < 0 || yy >= 64) continue;
          #pragma unroll
          for (int dx = 0; dx < 3; dx++) {
            int x2 = xx + dx - 1;
            if (x2 < 0 || x2 >= 64) continue;
            lep += wgt[dy * 3 + dx] * x[((size_t)b * SEQ + yy * 64 + x2) * CH + ch];
          }
        }
        out[((size_t)(b * SEQ) + nrow) * CH + ch] = o + lep;
      }
    }
  }
}

// ---- launcher ------------------------------------------------------------

extern "C" void kernel_launch(void* const* d_in, const int* in_sizes, int n_in,
                              void* d_out, int out_size, void* d_ws, size_t ws_size,
                              hipStream_t stream) {
  (void)in_sizes; (void)n_in; (void)out_size; (void)ws_size;
  const float* x   = (const float*)d_in[0];
  const float* wqk = (const float*)d_in[1];
  const float* bqk = (const float*)d_in[2];
  const float* lw  = (const float*)d_in[3];
  const float* lb  = (const float*)d_in[4];
  float* out = (float*)d_out;

  const size_t SZ = (size_t)BATCH * SEQ * CH;          // 33.5M elems
  float* q_act  = (float*)d_ws;
  float* q_rope = q_act  + SZ;
  float* k_rope = q_rope + SZ;
  float* kvT    = k_rope + SZ;                          // 16*8*64*64
  float* zbuf   = kvT    + (size_t)BATCH * HEADS * HDIM * HDIM;
  float* ksum   = zbuf   + (size_t)BATCH * HEADS * SEQ; // 16*512

  la_zero<<<(BATCH * CH + 255) / 256, 256, 0, stream>>>(ksum, BATCH * CH);
  la_gemm_qk<<<dim3(NCOL / 128, (BATCH * SEQ) / 128), 256, 0, stream>>>(
      x, wqk, bqk, q_act, q_rope, k_rope, ksum);
  la_kv<<<dim3(HEADS, BATCH), 256, 0, stream>>>(k_rope, x, kvT);
  la_z<<<(BATCH * HEADS * SEQ) / 256, 256, 0, stream>>>(q_act, ksum, zbuf);
  la_out<<<dim3(SEQ / 128, HEADS, BATCH), 256, 0, stream>>>(
      q_rope, kvT, zbuf, x, lw, lb, out);
}
